// GCN_55783035240587
// MI455X (gfx1250) — matrix operand order repaired and checked by reading, not compiled, not run on hardware
//
#include <hip/hip_runtime.h>
#include <hip/hip_bf16.h>

#define N_NODES 100000
#define N_EDGES 500000
#define D_FEAT  512

typedef __attribute__((ext_vector_type(16))) __bf16 v16bf;
typedef __attribute__((ext_vector_type(8)))  __bf16 v8bf;
typedef __attribute__((ext_vector_type(4)))  __bf16 v4bf;
typedef __attribute__((ext_vector_type(8)))  float  v8f;

// ---------------------------------------------------------------------------
// Degree / normalization kernels
// ---------------------------------------------------------------------------
__global__ void deg_init_kernel(float* __restrict__ deg) {
    int i = blockIdx.x * blockDim.x + threadIdx.x;
    if (i < N_NODES) deg[i] = 1.0f;   // self-loop weight
}

__global__ void deg_edge_kernel(const int* __restrict__ dst,
                                const float* __restrict__ w,
                                float* __restrict__ deg) {
    int e = blockIdx.x * blockDim.x + threadIdx.x;
    if (e < N_EDGES) atomicAdd(&deg[dst[e]], w[e]);
}

__global__ void dis_kernel(const float* __restrict__ deg, float* __restrict__ dis) {
    int i = blockIdx.x * blockDim.x + threadIdx.x;
    if (i < N_NODES) {
        float d = deg[i];
        dis[i] = d > 0.0f ? rsqrtf(d) : 0.0f;
    }
}

__global__ void norm_kernel(const int* __restrict__ src, const int* __restrict__ dst,
                            const float* __restrict__ w, const float* __restrict__ dis,
                            float* __restrict__ nrm) {
    int e = blockIdx.x * blockDim.x + threadIdx.x;
    if (e < N_EDGES) nrm[e] = dis[src[e]] * w[e] * dis[dst[e]];
}

// ---------------------------------------------------------------------------
// Split helpers for bf16x3 (hi + lo decomposition of f32)
// ---------------------------------------------------------------------------
__device__ __forceinline__ __bf16 bf_hi(float f) { return (__bf16)f; }
__device__ __forceinline__ __bf16 bf_lo(float f, __bf16 hi) {
    return (__bf16)(f - (float)hi);
}

// ---------------------------------------------------------------------------
// One-time W split: W[k][n] f32  ->  Whi_t[n][k], Wlo_t[n][k] bf16
// Transposed layout matches the WMMA B-fragment order (K contiguous per col),
// so B fragments load straight from global (W stays L2/WGP$ resident).
// ---------------------------------------------------------------------------
__global__ void w_split_kernel(const float* __restrict__ W,
                               __bf16* __restrict__ whi, __bf16* __restrict__ wlo) {
    const int idx = blockIdx.x * blockDim.x + threadIdx.x;  // 0 .. 512*512-1
    const int n = idx >> 9;
    const int k = idx & 511;
    const float f = W[(size_t)k * D_FEAT + n];
    const __bf16 h = bf_hi(f);
    whi[(size_t)n * D_FEAT + k] = h;
    wlo[(size_t)n * D_FEAT + k] = bf_lo(f, h);
}

// ---------------------------------------------------------------------------
// GEMM: xp = x @ W via split-bf16 WMMA (hi*hi + hi*lo + lo*hi, f32 accum)
// Block tile 32(M) x 256(N), 256 threads = 8 waves as 2(M) x 4(N);
// each wave owns 4 accumulator tiles (16x64) -> 12 WMMAs per K-chunk,
// A fragment reused 4x, 4 independent accumulation chains.
// Grid (2, 3125): column block fastest-varying -> x read from HBM once.
// Epilogue: store xp, and out = xp * dis^2 (fused self-loop message).
// ---------------------------------------------------------------------------
__global__ __launch_bounds__(256)
void gemm_bf16x3_kernel(const float* __restrict__ x,
                        const __bf16* __restrict__ whi, const __bf16* __restrict__ wlo,
                        const float* __restrict__ dis,
                        float* __restrict__ xp, float* __restrict__ out) {
    // A tile only (hi/lo), padded row stride (40 bf16 = 80B, multiple of 16B)
    __shared__ __bf16 Ahi[32][40];
    __shared__ __bf16 Alo[32][40];

    const int tid    = threadIdx.x;
    const int waveId = tid >> 5;
    const int lane   = tid & 31;
    const int msub   = waveId & 1;        // 0..1  (16-row subtile)
    const int nq     = waveId >> 1;       // 0..3  (64-col group)
    const int half   = lane >> 4;         // 0..1
    const int l16    = lane & 15;

    const int rowBase = blockIdx.y * 32;            // 3125 * 32 = 100000
    const int colBase = blockIdx.x * 256 + nq * 64; // 2 * 256 = 512

    v8f acc[4];
    #pragma unroll
    for (int i = 0; i < 4; ++i) acc[i] = (v8f){0.f,0.f,0.f,0.f,0.f,0.f,0.f,0.f};

    // A-tile load mapping: 32x32 f32, one float4 per thread
    const int arow = tid >> 3;            // 0..31
    const int akc  = (tid & 7) * 4;       // 0,4,..,28
    const int am   = msub * 16 + l16;     // fragment row in A tile

    for (int kBase = 0; kBase < D_FEAT; kBase += 32) {
        __syncthreads();

        // ---- stage A (x tile), split into hi/lo bf16 ----
        {
            const float4 v = *(const float4*)(x + (size_t)(rowBase + arow) * D_FEAT + kBase + akc);
            const __bf16 h0 = bf_hi(v.x), l0 = bf_lo(v.x, h0);
            const __bf16 h1 = bf_hi(v.y), l1 = bf_lo(v.y, h1);
            const __bf16 h2 = bf_hi(v.z), l2 = bf_lo(v.z, h2);
            const __bf16 h3 = bf_hi(v.w), l3 = bf_lo(v.w, h3);
            v4bf h = {h0, h1, h2, h3};
            v4bf l = {l0, l1, l2, l3};
            *(v4bf*)(&Ahi[arow][akc]) = h;
            *(v4bf*)(&Alo[arow][akc]) = l;
        }

        __syncthreads();

        // ---- A fragments (ISA layout: K in [8h,8h+8) ++ [16+8h,16+8h+8)) ----
        v8bf a0h = *(const v8bf*)(&Ahi[am][8 * half]);
        v8bf a1h = *(const v8bf*)(&Ahi[am][16 + 8 * half]);
        v8bf a0l = *(const v8bf*)(&Alo[am][8 * half]);
        v8bf a1l = *(const v8bf*)(&Alo[am][16 + 8 * half]);
        v16bf ahi = __builtin_shufflevector(a0h, a1h, 0,1,2,3,4,5,6,7,8,9,10,11,12,13,14,15);
        v16bf alo = __builtin_shufflevector(a0l, a1l, 0,1,2,3,4,5,6,7,8,9,10,11,12,13,14,15);

        // ---- 4 B-fragment tiles straight from global (L2-resident W) ----
        #pragma unroll
        for (int nt = 0; nt < 4; ++nt) {
            const int n = colBase + nt * 16 + l16;
            const size_t off = (size_t)n * D_FEAT + kBase + 8 * half;
            v8bf b0h = *(const v8bf*)(whi + off);
            v8bf b1h = *(const v8bf*)(whi + off + 16);
            v8bf b0l = *(const v8bf*)(wlo + off);
            v8bf b1l = *(const v8bf*)(wlo + off + 16);
            v16bf bhi = __builtin_shufflevector(b0h, b1h, 0,1,2,3,4,5,6,7,8,9,10,11,12,13,14,15);
            v16bf blo = __builtin_shufflevector(b0l, b1l, 0,1,2,3,4,5,6,7,8,9,10,11,12,13,14,15);

            acc[nt] = __builtin_amdgcn_wmma_f32_16x16x32_bf16(false, ahi, false, bhi,
                                                              (short)0, acc[nt], false, false);
            acc[nt] = __builtin_amdgcn_wmma_f32_16x16x32_bf16(false, ahi, false, blo,
                                                              (short)0, acc[nt], false, false);
            acc[nt] = __builtin_amdgcn_wmma_f32_16x16x32_bf16(false, alo, false, bhi,
                                                              (short)0, acc[nt], false, false);
        }
    }

    // ---- epilogue: xp store + fused self-loop message out = xp * dis^2 ----
    const int rtile = rowBase + msub * 16 + 8 * half;
    #pragma unroll
    for (int nt = 0; nt < 4; ++nt) {
        const int col = colBase + nt * 16 + l16;
        #pragma unroll
        for (int v = 0; v < 8; ++v) {
            const int r = rtile + v;
            const float val = acc[nt][v];
            const float s = dis[r];
            xp[(size_t)r * D_FEAT + col]  = val;
            out[(size_t)r * D_FEAT + col] = val * s * s;
        }
    }
}

// ---------------------------------------------------------------------------
// Edge scatter: out[dst] += xp[src] * norm.  2 edges per block, 128 threads
// per edge, float4 per thread (4 x global_atomic_add_f32).
// ---------------------------------------------------------------------------
__global__ __launch_bounds__(256)
void scatter_kernel(const float* __restrict__ xp, const float* __restrict__ nrm,
                    const int* __restrict__ src, const int* __restrict__ dst,
                    float* __restrict__ out) {
    const int sub = threadIdx.x >> 7;            // edge within block
    const int t   = threadIdx.x & 127;           // column group
    const int e   = blockIdx.x * 2 + sub;
    const int s   = src[e];
    const int d   = dst[e];
    const float n = nrm[e];
    const int col = t * 4;
    const float4 v = *(const float4*)(xp + (size_t)s * D_FEAT + col);
    float* o = out + (size_t)d * D_FEAT + col;
    atomicAdd(o + 0, v.x * n);
    atomicAdd(o + 1, v.y * n);
    atomicAdd(o + 2, v.z * n);
    atomicAdd(o + 3, v.w * n);
}

// ---------------------------------------------------------------------------
// out = relu(out + b), float4 per thread, in place on d_out
// ---------------------------------------------------------------------------
__global__ __launch_bounds__(256)
void bias_relu_kernel(const float* __restrict__ b, float* __restrict__ out) {
    const size_t idx = (size_t)blockIdx.x * blockDim.x + threadIdx.x;  // float4 index
    const int col = (int)((idx * 4) & (D_FEAT - 1));
    const float4 bb = *(const float4*)(b + col);
    float4 v = *(float4*)(out + idx * 4);
    v.x = fmaxf(v.x + bb.x, 0.f);
    v.y = fmaxf(v.y + bb.y, 0.f);
    v.z = fmaxf(v.z + bb.z, 0.f);
    v.w = fmaxf(v.w + bb.w, 0.f);
    *(float4*)(out + idx * 4) = v;
}

// ---------------------------------------------------------------------------
extern "C" void kernel_launch(void* const* d_in, const int* in_sizes, int n_in,
                              void* d_out, int out_size, void* d_ws, size_t ws_size,
                              hipStream_t stream) {
    const float* x  = (const float*)d_in[0];               // [N, 512]
    const float* ew = (const float*)d_in[1];               // [E]
    const float* W  = (const float*)d_in[2];               // [512, 512]
    const float* b  = (const float*)d_in[3];               // [512]
    const int*   ei = (const int*)d_in[4];                 // [2, E]
    const int* src = ei;
    const int* dst = ei + N_EDGES;

    float* out = (float*)d_out;

    // Workspace: xp [N*512] f32 | deg [N] | dis [N] | norm [E] | Whi_t, Wlo_t bf16
    float* xp  = (float*)d_ws;
    float* deg = xp  + (size_t)N_NODES * D_FEAT;
    float* dis = deg + N_NODES;
    float* nrm = dis + N_NODES;
    __bf16* whi = (__bf16*)(nrm + N_EDGES);
    __bf16* wlo = whi + (size_t)D_FEAT * D_FEAT;

    const int TB = 256;
    // 1) deg = 1 (self loops)
    deg_init_kernel<<<(N_NODES + TB - 1) / TB, TB, 0, stream>>>(deg);
    // 2) deg[dst] += w
    deg_edge_kernel<<<(N_EDGES + TB - 1) / TB, TB, 0, stream>>>(dst, ew, deg);
    // 3) dis = rsqrt(deg)
    dis_kernel<<<(N_NODES + TB - 1) / TB, TB, 0, stream>>>(deg, dis);
    // 4) per-edge norm
    norm_kernel<<<(N_EDGES + TB - 1) / TB, TB, 0, stream>>>(src, dst, ew, dis, nrm);
    // 5) one-time W hi/lo split into transposed bf16 layout
    w_split_kernel<<<(D_FEAT * D_FEAT) / TB, TB, 0, stream>>>(W, whi, wlo);
    // 6) xp = x @ W (split-bf16 WMMA) + fused self-loop init of out
    dim3 ggrid(D_FEAT / 256, N_NODES / 32, 1);             // (2, 3125), col fastest
    gemm_bf16x3_kernel<<<ggrid, TB, 0, stream>>>(x, whi, wlo, dis, xp, out);
    // 7) scatter messages
    scatter_kernel<<<N_EDGES / 2, TB, 0, stream>>>(xp, nrm, src, dst, out);
    // 8) bias + relu
    bias_relu_kernel<<<((size_t)N_NODES * D_FEAT / 4) / TB, TB, 0, stream>>>(b, out);
}
